// LSTM_Encoder_19078244729405
// MI455X (gfx1250) — compile-verified, matrix operand order checked by reading, not run on hardware
//
#include <hip/hip_runtime.h>

#define BB   64
#define SS   1024
#define DD   512
#define HH   1024
#define GG   4096
#define NWG  32
#define HC   32            // H columns per workgroup
#define LDSK 2056          // A-row stride in LDS (2048 + 8 pad halves)

typedef __bf16 bf16t;
typedef bf16t v16bf __attribute__((ext_vector_type(16)));
typedef float v8f   __attribute__((ext_vector_type(8)));

union Frag { v16bf v; uint4 q[2]; };

__device__ __forceinline__ unsigned short f2bf(float f) {
    union { float f; unsigned u; } v; v.f = f;
    unsigned r = v.u + 0x7fffu + ((v.u >> 16) & 1u);   // round-to-nearest-even
    return (unsigned short)(r >> 16);
}

__device__ __forceinline__ float sigf(float x) { return 1.0f / (1.0f + __expf(-x)); }

// ---------------------------------------------------------------------------
// Pack fp32 weight [G, Kw] into bf16 WMMA B-fragments:
//   frag (nt, kt) = 512 halves, lane-major: half idx = lane*16 + j
//   element = W[nt*16 + lane%16][kt*32 + (j/8)*16 + (lane/16)*8 + j%8]
// ---------------------------------------------------------------------------
__global__ void pack_weight(const float* __restrict__ W, unsigned short* __restrict__ out,
                            int Kw, int nkt, int total) {
    int idx = blockIdx.x * 256 + threadIdx.x;
    if (idx >= total) return;
    int f    = idx >> 9;
    int w    = idx & 511;
    int lane = w >> 4;
    int j    = w & 15;
    int nt   = f / nkt;
    int kt   = f - nt * nkt;
    int g    = nt * 16 + (lane & 15);
    int k    = kt * 32 + ((j >> 3) << 4) + ((lane >> 4) << 3) + (j & 7);
    out[idx] = f2bf(W[(size_t)g * Kw + k]);
}

__global__ void make_bias(const float* bi0, const float* bh0,
                          const float* bi1, const float* bh1,
                          float* bias0, float* bias1) {
    int i = blockIdx.x * 256 + threadIdx.x;
    if (i < GG) { bias0[i] = bi0[i] + bh0[i]; bias1[i] = bi1[i] + bh1[i]; }
}

__global__ void init_state(unsigned* ctr, unsigned* h0, unsigned* h1, int nwords) {
    int i = blockIdx.x * 256 + threadIdx.x;
    if (i == 0) *ctr = 0;
    if (i < nwords) { h0[i] = 0u; h1[i] = 0u; }
}

// ---------------------------------------------------------------------------
// Device-wide barrier: one per timestep.  ctr is zeroed each launch, targets
// are monotonically increasing within a launch -> deterministic across replays.
// ---------------------------------------------------------------------------
__device__ __forceinline__ void grid_barrier(unsigned* ctr, unsigned target) {
    __threadfence();
    __syncthreads();
    if (threadIdx.x == 0) {
        __hip_atomic_fetch_add(ctr, 1u, __ATOMIC_RELEASE, __HIP_MEMORY_SCOPE_AGENT);
        while (__hip_atomic_load(ctr, __ATOMIC_ACQUIRE, __HIP_MEMORY_SCOPE_AGENT) < target) {
            __builtin_amdgcn_s_sleep(1);
        }
    }
    __syncthreads();
    __threadfence();
}

// ---------------------------------------------------------------------------
// Software-pipelined K-loop: prefetch A (LDS) and B (global packed) fragments
// for kt+1 while the four WMMAs for kt issue.  unroll 2 keeps the ping-pong
// buffer index compile-time constant (fragments stay in VGPRs).
// ---------------------------------------------------------------------------
__device__ __forceinline__ void gemm_block(const unsigned short* aLds,
                                           const unsigned short* __restrict__ Wp,
                                           int ktBegin, int ktEnd, int ktWOff, int nktW,
                                           int ntg, int lane, v8f acc[4]) {
    const int lrow = lane & 15;
    const int lgrp = lane >> 4;
    Frag af[2][4];
    Frag bfr[2];

    {   // prologue: fetch fragments for ktBegin
        const unsigned short* bp =
            Wp + (((size_t)(ntg * nktW + (ktBegin - ktWOff))) << 9) + lane * 16;
        bfr[0].q[0] = *(const uint4*)bp;
        bfr[0].q[1] = *(const uint4*)(bp + 8);
#pragma unroll
        for (int mt = 0; mt < 4; ++mt) {
            const int base = (mt * 16 + lrow) * LDSK + ktBegin * 32 + lgrp * 8;
            af[0][mt].q[0] = *(const uint4*)&aLds[base];
            af[0][mt].q[1] = *(const uint4*)&aLds[base + 16];
        }
    }

#pragma unroll 2
    for (int kt = ktBegin; kt < ktEnd; ++kt) {
        const int cur = (kt - ktBegin) & 1;
        const int nxt = cur ^ 1;
        if (kt + 1 < ktEnd) {   // prefetch kt+1
            const unsigned short* bp =
                Wp + (((size_t)(ntg * nktW + (kt + 1 - ktWOff))) << 9) + lane * 16;
            bfr[nxt].q[0] = *(const uint4*)bp;
            bfr[nxt].q[1] = *(const uint4*)(bp + 8);
#pragma unroll
            for (int mt = 0; mt < 4; ++mt) {
                const int base = (mt * 16 + lrow) * LDSK + (kt + 1) * 32 + lgrp * 8;
                af[nxt][mt].q[0] = *(const uint4*)&aLds[base];
                af[nxt][mt].q[1] = *(const uint4*)&aLds[base + 16];
            }
        }
#pragma unroll
        for (int mt = 0; mt < 4; ++mt) {
            acc[mt] = __builtin_amdgcn_wmma_f32_16x16x32_bf16(
                false, af[cur][mt].v, false, bfr[cur].v, (short)0, acc[mt], false, false);
        }
    }
}

// ---------------------------------------------------------------------------
// Persistent 2-layer LSTM kernel.  32 WGs x 256 threads (8 wave32).
// ---------------------------------------------------------------------------
__global__ void __launch_bounds__(256, 1)
lstm_persistent(const float* __restrict__ x,
                const unsigned short* __restrict__ Wih0p,
                const unsigned short* __restrict__ Whh0p,
                const unsigned short* __restrict__ Wih1p,
                const unsigned short* __restrict__ Whh1p,
                const float* __restrict__ bias0,
                const float* __restrict__ bias1,
                unsigned short* h0buf, unsigned short* h1buf,
                unsigned* ctr, float* out) {
    extern __shared__ char smem[];
    unsigned short* aLds = (unsigned short*)smem;                 // [64][LDSK] bf16
    float* gbuf = (float*)(smem + (size_t)BB * LDSK * 2);         // [4][64][HC] f32

    const int tid  = threadIdx.x;
    const int wg   = blockIdx.x;
    const int wave = tid >> 5;
    const int lane = tid & 31;
    const int q    = wave >> 1;          // gate 0..3 (i,f,g,o)
    const int ntl  = wave & 1;           // n-subtile within WG slice
    const int ntg  = q * (HH / 16) + wg * 2 + ntl;   // global 16-col tile in G
    const int lrow = lane & 15;
    const int lgrp = lane >> 4;

    const float bv0 = bias0[ntg * 16 + lrow];
    const float bv1 = bias1[ntg * 16 + lrow];

    // Elementwise mapping: 8 consecutive columns per thread (vectorizable).
    const int eb  = tid >> 2;            // batch row 0..63
    const int en0 = (tid & 3) * 8;       // first of 8 consecutive columns

    float c0s[8], c1s[8];
#pragma unroll
    for (int u = 0; u < 8; ++u) { c0s[u] = 0.0f; c1s[u] = 0.0f; }

    const size_t coutO = (size_t)BB * SS * HH;
    const size_t hfinO = 2 * (size_t)BB * SS * HH;
    const size_t cfinO = hfinO + (size_t)BB * HH;

    for (int t = 0; t < SS; ++t) {
        const int p  = t & 1;
        const int pp = p ^ 1;

        // ---- stage layer0 A into LDS: x_t -> cols [0,512), h0_prev -> [512,1536)
        for (int e = tid; e < (BB * DD) / 4; e += 256) {
            int k4 = e << 2;
            int b = k4 >> 9, k = k4 & (DD - 1);
            const float4 xv = *(const float4*)&x[(size_t)b * SS * DD + (size_t)t * DD + k];
            unsigned lo = (unsigned)f2bf(xv.x) | ((unsigned)f2bf(xv.y) << 16);
            unsigned hi = (unsigned)f2bf(xv.z) | ((unsigned)f2bf(xv.w) << 16);
            *(uint2*)&aLds[b * LDSK + k] = make_uint2(lo, hi);
        }
        {
            const uint4* src = (const uint4*)(h0buf + (size_t)pp * BB * HH);
            for (int e = tid; e < (BB * HH) / 8; e += 256) {
                int f8 = e << 3;
                int b = f8 >> 10, hk = f8 & (HH - 1);
                *(uint4*)&aLds[b * LDSK + DD + hk] = src[e];
            }
        }
        __syncthreads();

        // ---- layer0 GEMM: gates = [x|h0] * [Wih0|Whh0]^T + bias0
        v8f acc[4];
#pragma unroll
        for (int mt = 0; mt < 4; ++mt)
#pragma unroll
            for (int r = 0; r < 8; ++r) acc[mt][r] = bv0;

        gemm_block(aLds, Wih0p, 0, 16, 0, DD / 32, ntg, lane, acc);
        gemm_block(aLds, Whh0p, 16, 48, 16, HH / 32, ntg, lane, acc);

#pragma unroll
        for (int mt = 0; mt < 4; ++mt)
#pragma unroll
            for (int r = 0; r < 8; ++r) {
                int m = mt * 16 + r + 8 * lgrp;
                gbuf[(q * BB + m) * HC + ntl * 16 + lrow] = acc[mt][r];
            }
        __syncthreads();

        // ---- layer0 elementwise (c0 in registers; vector LDS reads, uint4 h store)
        {
            float4 gi0 = *(const float4*)&gbuf[(0 * BB + eb) * HC + en0];
            float4 gi1 = *(const float4*)&gbuf[(0 * BB + eb) * HC + en0 + 4];
            float4 gf0 = *(const float4*)&gbuf[(1 * BB + eb) * HC + en0];
            float4 gf1 = *(const float4*)&gbuf[(1 * BB + eb) * HC + en0 + 4];
            float4 gg0 = *(const float4*)&gbuf[(2 * BB + eb) * HC + en0];
            float4 gg1 = *(const float4*)&gbuf[(2 * BB + eb) * HC + en0 + 4];
            float4 go0 = *(const float4*)&gbuf[(3 * BB + eb) * HC + en0];
            float4 go1 = *(const float4*)&gbuf[(3 * BB + eb) * HC + en0 + 4];
            float gi[8] = {gi0.x, gi0.y, gi0.z, gi0.w, gi1.x, gi1.y, gi1.z, gi1.w};
            float gf[8] = {gf0.x, gf0.y, gf0.z, gf0.w, gf1.x, gf1.y, gf1.z, gf1.w};
            float gg[8] = {gg0.x, gg0.y, gg0.z, gg0.w, gg1.x, gg1.y, gg1.z, gg1.w};
            float go[8] = {go0.x, go0.y, go0.z, go0.w, go1.x, go1.y, go1.z, go1.w};
            unsigned hp[4];
#pragma unroll
            for (int u = 0; u < 8; ++u) {
                float c = sigf(gf[u]) * c0s[u] + sigf(gi[u]) * tanhf(gg[u]);
                c0s[u] = c;
                float h = sigf(go[u]) * tanhf(c);
                unsigned short hb = f2bf(h);
                if (u & 1) hp[u >> 1] |= ((unsigned)hb << 16);
                else       hp[u >> 1] = hb;
            }
            *(uint4*)&h0buf[(size_t)p * BB * HH + eb * HH + wg * HC + en0] =
                make_uint4(hp[0], hp[1], hp[2], hp[3]);
        }

        // ---- the one device-wide barrier per step
        grid_barrier(ctr, (unsigned)(t + 1) * NWG);

        // ---- stage layer1 A: h0_cur -> cols [0,1024), h1_prev -> [1024,2048)
        {
            const uint4* s0 = (const uint4*)(h0buf + (size_t)p  * BB * HH);
            const uint4* s1 = (const uint4*)(h1buf + (size_t)pp * BB * HH);
            for (int e = tid; e < (BB * HH) / 8; e += 256) {
                int f8 = e << 3;
                int b = f8 >> 10, hk = f8 & (HH - 1);
                *(uint4*)&aLds[b * LDSK + hk]      = s0[e];
                *(uint4*)&aLds[b * LDSK + HH + hk] = s1[e];
            }
        }
        __syncthreads();

        // ---- layer1 GEMM
#pragma unroll
        for (int mt = 0; mt < 4; ++mt)
#pragma unroll
            for (int r = 0; r < 8; ++r) acc[mt][r] = bv1;

        gemm_block(aLds, Wih1p, 0, 32, 0, HH / 32, ntg, lane, acc);
        gemm_block(aLds, Whh1p, 32, 64, 32, HH / 32, ntg, lane, acc);

#pragma unroll
        for (int mt = 0; mt < 4; ++mt)
#pragma unroll
            for (int r = 0; r < 8; ++r) {
                int m = mt * 16 + r + 8 * lgrp;
                gbuf[(q * BB + m) * HC + ntl * 16 + lrow] = acc[mt][r];
            }
        __syncthreads();

        // ---- layer1 elementwise + outputs (vector reads/writes)
        {
            float4 gi0 = *(const float4*)&gbuf[(0 * BB + eb) * HC + en0];
            float4 gi1 = *(const float4*)&gbuf[(0 * BB + eb) * HC + en0 + 4];
            float4 gf0 = *(const float4*)&gbuf[(1 * BB + eb) * HC + en0];
            float4 gf1 = *(const float4*)&gbuf[(1 * BB + eb) * HC + en0 + 4];
            float4 gg0 = *(const float4*)&gbuf[(2 * BB + eb) * HC + en0];
            float4 gg1 = *(const float4*)&gbuf[(2 * BB + eb) * HC + en0 + 4];
            float4 go0 = *(const float4*)&gbuf[(3 * BB + eb) * HC + en0];
            float4 go1 = *(const float4*)&gbuf[(3 * BB + eb) * HC + en0 + 4];
            float gi[8] = {gi0.x, gi0.y, gi0.z, gi0.w, gi1.x, gi1.y, gi1.z, gi1.w};
            float gf[8] = {gf0.x, gf0.y, gf0.z, gf0.w, gf1.x, gf1.y, gf1.z, gf1.w};
            float gg[8] = {gg0.x, gg0.y, gg0.z, gg0.w, gg1.x, gg1.y, gg1.z, gg1.w};
            float go[8] = {go0.x, go0.y, go0.z, go0.w, go1.x, go1.y, go1.z, go1.w};
            float hv[8], cv[8];
            unsigned hp[4];
#pragma unroll
            for (int u = 0; u < 8; ++u) {
                float c = sigf(gf[u]) * c1s[u] + sigf(gi[u]) * tanhf(gg[u]);
                c1s[u] = c;
                float h = sigf(go[u]) * tanhf(c);
                hv[u] = h; cv[u] = c;
                unsigned short hb = f2bf(h);
                if (u & 1) hp[u >> 1] |= ((unsigned)hb << 16);
                else       hp[u >> 1] = hb;
            }
            const int hcol0 = wg * HC + en0;
            *(uint4*)&h1buf[(size_t)p * BB * HH + eb * HH + hcol0] =
                make_uint4(hp[0], hp[1], hp[2], hp[3]);
            float* po = out + (size_t)eb * SS * HH + (size_t)t * HH + hcol0;
            *(float4*)po       = make_float4(hv[0], hv[1], hv[2], hv[3]);
            *(float4*)(po + 4) = make_float4(hv[4], hv[5], hv[6], hv[7]);
            float* pc = out + coutO + (size_t)eb * SS * HH + (size_t)t * HH + hcol0;
            *(float4*)pc       = make_float4(cv[0], cv[1], cv[2], cv[3]);
            *(float4*)(pc + 4) = make_float4(cv[4], cv[5], cv[6], cv[7]);
            if (t == SS - 1) {
                float* ph = out + hfinO + (size_t)eb * HH + hcol0;
                *(float4*)ph       = make_float4(hv[0], hv[1], hv[2], hv[3]);
                *(float4*)(ph + 4) = make_float4(hv[4], hv[5], hv[6], hv[7]);
                float* pf = out + cfinO + (size_t)eb * HH + hcol0;
                *(float4*)pf       = make_float4(cv[0], cv[1], cv[2], cv[3]);
                *(float4*)(pf + 4) = make_float4(cv[4], cv[5], cv[6], cv[7]);
            }
        }
        __syncthreads();   // protect aLds/gbuf reuse at next loop top
    }
}

// ---------------------------------------------------------------------------
extern "C" void kernel_launch(void* const* d_in, const int* in_sizes, int n_in,
                              void* d_out, int out_size, void* d_ws, size_t ws_size,
                              hipStream_t stream) {
    (void)in_sizes; (void)n_in; (void)out_size; (void)ws_size;

    const float* x    = (const float*)d_in[0];
    const float* Wih0 = (const float*)d_in[1];
    const float* Whh0 = (const float*)d_in[2];
    const float* bih0 = (const float*)d_in[3];
    const float* bhh0 = (const float*)d_in[4];
    const float* Wih1 = (const float*)d_in[5];
    const float* Whh1 = (const float*)d_in[6];
    const float* bih1 = (const float*)d_in[7];
    const float* bhh1 = (const float*)d_in[8];

    char* ws = (char*)d_ws;
    size_t off = 0;
    auto carve = [&](size_t bytes) -> void* {
        void* ptr = ws + off;
        off = (off + bytes + 255) & ~(size_t)255;
        return ptr;
    };

    unsigned*       ctr   = (unsigned*)carve(sizeof(unsigned));
    unsigned short* Wih0p = (unsigned short*)carve(2ull * GG * DD);
    unsigned short* Whh0p = (unsigned short*)carve(2ull * GG * HH);
    unsigned short* Wih1p = (unsigned short*)carve(2ull * GG * HH);
    unsigned short* Whh1p = (unsigned short*)carve(2ull * GG * HH);
    float*          bias0 = (float*)carve(4ull * GG);
    float*          bias1 = (float*)carve(4ull * GG);
    unsigned short* h0buf = (unsigned short*)carve(2ull * 2 * BB * HH);
    unsigned short* h1buf = (unsigned short*)carve(2ull * 2 * BB * HH);

    // One-time (per launch) weight/bias packing and state init — stream ordered.
    pack_weight<<<(GG * DD + 255) / 256, 256, 0, stream>>>(Wih0, Wih0p, DD, DD / 32, GG * DD);
    pack_weight<<<(GG * HH + 255) / 256, 256, 0, stream>>>(Whh0, Whh0p, HH, HH / 32, GG * HH);
    pack_weight<<<(GG * HH + 255) / 256, 256, 0, stream>>>(Wih1, Wih1p, HH, HH / 32, GG * HH);
    pack_weight<<<(GG * HH + 255) / 256, 256, 0, stream>>>(Whh1, Whh1p, HH, HH / 32, GG * HH);
    make_bias<<<(GG + 255) / 256, 256, 0, stream>>>(bih0, bhh0, bih1, bhh1, bias0, bias1);

    const int nwords = (2 * BB * HH * 2) / 4;   // words per h buffer pair
    init_state<<<(nwords + 255) / 256, 256, 0, stream>>>(ctr, (unsigned*)h0buf, (unsigned*)h1buf, nwords);

    // Persistent recurrent kernel: 32 WGs, 8 wave32 each, ~289 KB LDS.
    const size_t smemBytes = (size_t)BB * LDSK * 2 + (size_t)4 * BB * HC * 4;
    lstm_persistent<<<NWG, 256, smemBytes, stream>>>(
        x, Wih0p, Whh0p, Wih1p, Whh1p, bias0, bias1, h0buf, h1buf, ctr, (float*)d_out);
}
